// GMMILDiscriminator_52355651338497
// MI455X (gfx1250) — matrix operand order repaired and checked by reading, not compiled, not run on hardware
//
#include <hip/hip_runtime.h>
#include <hip/hip_bf16.h>

typedef float v2f __attribute__((ext_vector_type(2)));
typedef float v8f __attribute__((ext_vector_type(8)));
typedef __bf16 v8bf  __attribute__((ext_vector_type(8)));
typedef __bf16 v16bf __attribute__((ext_vector_type(16)));

#define N1   4096
#define DIM  256
#define SDIM 192
#define ADIM 64

// ---------- helpers ----------
__device__ __forceinline__ unsigned int f2key(float f) {
  unsigned int u = __float_as_uint(f);
  return (u & 0x80000000u) ? ~u : (u | 0x80000000u);
}
__device__ __forceinline__ float key2f(unsigned int k) {
  unsigned int u = (k & 0x80000000u) ? (k ^ 0x80000000u) : ~k;
  return __uint_as_float(u);
}

// ---------- pack [state|action] row, emit f32 + bf16 hi/lo planes + row norm ----------
__global__ void __launch_bounds__(256)
pack_norm_kernel(const float* __restrict__ s, const float* __restrict__ a,
                 float* __restrict__ dst, __bf16* __restrict__ dsth,
                 __bf16* __restrict__ dstl, float* __restrict__ norm) {
  __shared__ float red[256];
  const int row = blockIdx.x, t = threadIdx.x;
  const size_t idx = (size_t)row * DIM + t;
  float v = (t < SDIM) ? s[(size_t)row * SDIM + t] : a[(size_t)row * ADIM + (t - SDIM)];
  dst[idx] = v;
  const __bf16 h = (__bf16)v;
  dsth[idx] = h;
  dstl[idx] = (__bf16)(v - (float)h);
  red[t] = v * v;
  __syncthreads();
  for (int off = 128; off > 0; off >>= 1) {
    if (t < off) red[t] += red[t + off];
    __syncthreads();
  }
  if (t == 0) norm[row] = red[0];
}

// ---------- column norms of esa (features as points for gamma_2) ----------
__global__ void __launch_bounds__(256)
colnorm_kernel(const float* __restrict__ E, float* __restrict__ c2) {
  __shared__ float red[256];
  const int f = blockIdx.x, t = threadIdx.x;
  float s = 0.0f;
  for (int n = t; n < N1; n += 256) {
    float v = E[(size_t)n * DIM + f];
    s += v * v;
  }
  red[t] = s;
  __syncthreads();
  for (int off = 128; off > 0; off >>= 1) {
    if (t < off) red[t] += red[t + off];
    __syncthreads();
  }
  if (t == 0) c2[f] = red[0];
}

// ---------- D1 = (x2 + y2 - 2*SA*ESA^T)/256  via bf16x3 split-precision WMMA ----------
// block = 256 thr = 8 waves (2 along M x 4 along N); block tile 128x128;
// wave tile 64x32 = 4x2 fragments of 16x16; K-step 32.
__global__ void __launch_bounds__(256)
dist_gemm_kernel(const __bf16* __restrict__ Ah, const __bf16* __restrict__ Al,
                 const __bf16* __restrict__ Bh, const __bf16* __restrict__ Bl,
                 const float* __restrict__ x2, const float* __restrict__ y2,
                 float* __restrict__ D) {
  const int lane = threadIdx.x & 31;
  const int widx = threadIdx.x >> 5;
  const int wm = widx >> 2, wn = widx & 3;
  const int m_block = blockIdx.y * 128, n_block = blockIdx.x * 128;
  const int lm = lane & 15;
  const int g  = lane >> 4;

  v8f acc[4][2] = {};

  const size_t aoff = (size_t)(m_block + wm * 64 + lm) * DIM;
  const size_t boff = (size_t)(n_block + wn * 32 + lm) * DIM;

  for (int k0 = 0; k0 < DIM; k0 += 32) {
    v16bf ah[4], al[4], bh[2], bl[2];
#pragma unroll
    for (int mi = 0; mi < 4; ++mi) {
      const size_t o = aoff + (size_t)mi * 16 * DIM + k0;
      ((v8bf*)&ah[mi])[0] = *(const v8bf*)(Ah + o + g * 8);
      ((v8bf*)&ah[mi])[1] = *(const v8bf*)(Ah + o + 16 + g * 8);
      ((v8bf*)&al[mi])[0] = *(const v8bf*)(Al + o + g * 8);
      ((v8bf*)&al[mi])[1] = *(const v8bf*)(Al + o + 16 + g * 8);
    }
#pragma unroll
    for (int ni = 0; ni < 2; ++ni) {
      const size_t o = boff + (size_t)ni * 16 * DIM + k0 + g * 8;
      ((v8bf*)&bh[ni])[0] = *(const v8bf*)(Bh + o);
      ((v8bf*)&bh[ni])[1] = *(const v8bf*)(Bh + o + 16);
      ((v8bf*)&bl[ni])[0] = *(const v8bf*)(Bl + o);
      ((v8bf*)&bl[ni])[1] = *(const v8bf*)(Bl + o + 16);
    }
#pragma unroll
    for (int mi = 0; mi < 4; ++mi)
#pragma unroll
      for (int ni = 0; ni < 2; ++ni) {
        acc[mi][ni] = __builtin_amdgcn_wmma_f32_16x16x32_bf16(
            false, ah[mi], false, bh[ni], (short)0, acc[mi][ni], false, false);
        acc[mi][ni] = __builtin_amdgcn_wmma_f32_16x16x32_bf16(
            false, ah[mi], false, bl[ni], (short)0, acc[mi][ni], false, false);
        acc[mi][ni] = __builtin_amdgcn_wmma_f32_16x16x32_bf16(
            false, al[mi], false, bh[ni], (short)0, acc[mi][ni], false, false);
      }
  }

  const int rbase = g * 8;
#pragma unroll
  for (int mi = 0; mi < 4; ++mi) {
#pragma unroll
    for (int ni = 0; ni < 2; ++ni) {
      const int n = n_block + wn * 32 + ni * 16 + lm;
      const float yn = y2[n];
#pragma unroll
      for (int r = 0; r < 8; ++r) {
        const int m = m_block + wm * 64 + mi * 16 + rbase + r;
        D[(size_t)m * N1 + n] = (x2[m] + yn - 2.0f * acc[mi][ni][r]) * (1.0f / 256.0f);
      }
    }
  }
}

// ---------- DDD = (c2_i + c2_j - 2*(ESA^T ESA))/4096 : 256x256, K=4096 (exact f32 WMMA) ----------
__global__ void __launch_bounds__(256)
gram_gemm_kernel(const float* __restrict__ E, const float* __restrict__ c2,
                 float* __restrict__ DD) {
  const int lane = threadIdx.x & 31;
  const int widx = threadIdx.x >> 5;
  const int w = blockIdx.x * 8 + widx;  // 0..255
  const int mt = w >> 4, nt = w & 15;
  const int lm = lane & 15;
  const int kh = (lane >> 4) * 2;
  const int m = mt * 16 + lm, n = nt * 16 + lm;

  v8f acc = {};
  for (int k0 = 0; k0 < N1; k0 += 4) {
    v2f a, b;
    a.x = E[(size_t)(k0 + kh) * DIM + m];
    a.y = E[(size_t)(k0 + kh + 1) * DIM + m];
    b.x = E[(size_t)(k0 + kh) * DIM + n];
    b.y = E[(size_t)(k0 + kh + 1) * DIM + n];
    acc = __builtin_amdgcn_wmma_f32_16x16x4_f32(false, a, false, b, (short)0, acc,
                                                false, false);
  }
  const int rbase = (lane >> 4) * 8;
  const float cn = c2[n];
#pragma unroll
  for (int r = 0; r < 8; ++r) {
    const int mm = mt * 16 + rbase + r;
    DD[(size_t)mm * 256 + n] = (c2[mm] + cn - 2.0f * acc[r]) * (1.0f / 4096.0f);
  }
}

// ---------- radix-select (exact lower median) machinery ----------
__global__ void sel_init_kernel(unsigned int* st1, unsigned int* st2,
                                unsigned int* hist, float* selfacc,
                                unsigned int k1, unsigned int k2) {
  const int t = blockIdx.x * blockDim.x + threadIdx.x;
  if (t == 0) { st1[0] = 0u; st1[1] = k1; st2[0] = 0u; st2[1] = k2; }
  if (t < 256) hist[t] = 0u;
  for (int i = t; i < N1; i += gridDim.x * blockDim.x) selfacc[i] = 0.0f;
}

__global__ void __launch_bounds__(256)
sel_count_kernel(const float* __restrict__ data, int n,
                 const unsigned int* __restrict__ st,
                 unsigned int* __restrict__ hist, int shift) {
  __shared__ unsigned int lh[256];
  lh[threadIdx.x] = 0u;
  __syncthreads();
  const unsigned int prefix = st[0];
  const int stride = gridDim.x * blockDim.x;
  for (int i = blockIdx.x * blockDim.x + threadIdx.x; i < n; i += stride) {
    const unsigned int key = f2key(data[i]);
    const unsigned int hi = (shift < 24) ? (key >> (shift + 8)) : 0u;
    if (shift == 24 || hi == prefix)
      atomicAdd(&lh[(key >> shift) & 255u], 1u);
  }
  __syncthreads();
  if (lh[threadIdx.x]) atomicAdd(&hist[threadIdx.x], lh[threadIdx.x]);
}

__global__ void sel_scan_kernel(unsigned int* st, unsigned int* hist) {
  __shared__ unsigned int h[256];
  h[threadIdx.x] = hist[threadIdx.x];
  hist[threadIdx.x] = 0u;  // reset for next round
  __syncthreads();
  if (threadIdx.x == 0) {
    const unsigned int k = st[1];
    unsigned int cum = 0, b = 0;
    for (b = 0; b < 256; ++b) {
      const unsigned int c = h[b];
      if (cum + c > k) break;
      cum += c;
    }
    if (b > 255u) b = 255u;
    st[0] = (st[0] << 8) | b;
    st[1] = k - cum;
  }
}

__global__ void gamma_kernel(const unsigned int* st1, const unsigned int* st2,
                             float* gammas) {
  if (threadIdx.x == 0) {
    gammas[0] = 1.0f / (key2f(st1[0]) + 1e-8f);
    gammas[1] = 1.0f / (key2f(st2[0]) + 1e-8f);
  }
}

// ---------- similarity: row sums of exp(-g*D1) ----------
__global__ void __launch_bounds__(256)
simil_kernel(const float* __restrict__ D, const float* __restrict__ gammas,
             float* __restrict__ sim) {
  __shared__ float red[256];
  const float g1 = gammas[0], g2 = gammas[1];
  const size_t row = blockIdx.x;
  float s = 0.0f;
  for (int e = threadIdx.x; e < N1; e += 256) {
    const float v = D[row * N1 + e];
    s += __expf(-g1 * v) + __expf(-g2 * v);
  }
  red[threadIdx.x] = s;
  __syncthreads();
  for (int off = 128; off > 0; off >>= 1) {
    if (threadIdx.x < off) red[threadIdx.x] += red[threadIdx.x + off];
    __syncthreads();
  }
  if (threadIdx.x == 0) sim[row] = red[0];
}

// ---------- fused self-GEMM (bf16x3) + exp column-sum (D_self never materialized) ----------
// 4096 waves; wave w: columns ntile*16..+15, rows mchunk*256..+255 (16 m-frags).
__global__ void __launch_bounds__(256)
self_gemm_kernel(const __bf16* __restrict__ Ah, const __bf16* __restrict__ Al,
                 const float* __restrict__ x2, const float* __restrict__ gammas,
                 float* __restrict__ selfacc) {
  const int lane = threadIdx.x & 31;
  const int widx = threadIdx.x >> 5;
  const int w = blockIdx.x * 8 + widx;  // 0..4095
  const int ntile = w & 255;
  const int mchunk = w >> 8;
  const int lm = lane & 15;
  const int g = lane >> 4;
  const float g1 = gammas[0], g2 = gammas[1];
  const int n = ntile * 16 + lm;
  const float xn = x2[n];
  const size_t boff = (size_t)n * DIM;
  const int rbase = g * 8;
  float s = 0.0f;

  for (int mt = 0; mt < 16; ++mt) {
    const int m0 = mchunk * 256 + mt * 16;
    const size_t aoff = (size_t)(m0 + lm) * DIM;
    v8f acc = {};
    for (int k0 = 0; k0 < DIM; k0 += 32) {
      v16bf ah, al, bh, bl;
      ((v8bf*)&ah)[0] = *(const v8bf*)(Ah + aoff + k0 + g * 8);
      ((v8bf*)&ah)[1] = *(const v8bf*)(Ah + aoff + k0 + 16 + g * 8);
      ((v8bf*)&al)[0] = *(const v8bf*)(Al + aoff + k0 + g * 8);
      ((v8bf*)&al)[1] = *(const v8bf*)(Al + aoff + k0 + 16 + g * 8);
      ((v8bf*)&bh)[0] = *(const v8bf*)(Ah + boff + k0 + g * 8);
      ((v8bf*)&bh)[1] = *(const v8bf*)(Ah + boff + k0 + 16 + g * 8);
      ((v8bf*)&bl)[0] = *(const v8bf*)(Al + boff + k0 + g * 8);
      ((v8bf*)&bl)[1] = *(const v8bf*)(Al + boff + k0 + 16 + g * 8);
      acc = __builtin_amdgcn_wmma_f32_16x16x32_bf16(false, ah, false, bh, (short)0,
                                                    acc, false, false);
      acc = __builtin_amdgcn_wmma_f32_16x16x32_bf16(false, ah, false, bl, (short)0,
                                                    acc, false, false);
      acc = __builtin_amdgcn_wmma_f32_16x16x32_bf16(false, al, false, bh, (short)0,
                                                    acc, false, false);
    }
#pragma unroll
    for (int r = 0; r < 8; ++r) {
      const int m = m0 + rbase + r;
      const float d = (x2[m] + xn - 2.0f * acc[r]) * (1.0f / 256.0f);
      s += __expf(-g1 * d) + __expf(-g2 * d);
    }
  }
  atomicAdd(&selfacc[n], s);
}

// ---------- output ----------
__global__ void final_kernel(const float* __restrict__ sim,
                             const float* __restrict__ selfacc,
                             float* __restrict__ out) {
  const int i = blockIdx.x * blockDim.x + threadIdx.x;
  if (i < N1) out[i] = (sim[i] - selfacc[i]) * (1.0f / 4096.0f);
}

extern "C" void kernel_launch(void* const* d_in, const int* in_sizes, int n_in,
                              void* d_out, int out_size, void* d_ws, size_t ws_size,
                              hipStream_t stream) {
  const float* state   = (const float*)d_in[0];
  const float* action  = (const float*)d_in[1];
  const float* estate  = (const float*)d_in[2];
  const float* eaction = (const float*)d_in[3];
  float* out = (float*)d_out;

  char* ws = (char*)d_ws;
  // layout (bytes); total ~82 MB
  float*  SA   = (float*)(ws + 0);                     //  4 MB
  float*  ESA  = (float*)(ws + (size_t)4  * 1048576);  //  4 MB
  float*  D1   = (float*)(ws + (size_t)8  * 1048576);  // 64 MB
  float*  DDD  = (float*)(ws + (size_t)72 * 1048576);  // 256 KB
  __bf16* SAH  = (__bf16*)(ws + (size_t)73 * 1048576); //  2 MB
  __bf16* SAL  = (__bf16*)(ws + (size_t)75 * 1048576); //  2 MB
  __bf16* ESAH = (__bf16*)(ws + (size_t)77 * 1048576); //  2 MB
  __bf16* ESAL = (__bf16*)(ws + (size_t)79 * 1048576); //  2 MB
  char*   tail = ws + (size_t)81 * 1048576;
  float* X2   = (float*)(tail + 0);        // 16 KB
  float* Y2   = (float*)(tail + 16384);    // 16 KB
  float* C2   = (float*)(tail + 32768);    //  1 KB
  float* SIM  = (float*)(tail + 34816);    // 16 KB
  float* SELF = (float*)(tail + 51200);    // 16 KB
  float* GAM  = (float*)(tail + 67584);
  unsigned int* HIST = (unsigned int*)(tail + 67840);
  unsigned int* ST1  = (unsigned int*)(tail + 68864);
  unsigned int* ST2  = (unsigned int*)(tail + 69120);

  // Phase A: pack + split + norms
  pack_norm_kernel<<<N1, 256, 0, stream>>>(state, action, SA, SAH, SAL, X2);
  pack_norm_kernel<<<N1, 256, 0, stream>>>(estate, eaction, ESA, ESAH, ESAL, Y2);
  colnorm_kernel<<<DIM, 256, 0, stream>>>(ESA, C2);

  // Phase B: distance GEMMs (bf16x3 WMMA big, f32 WMMA small)
  dist_gemm_kernel<<<dim3(32, 32), 256, 0, stream>>>(SAH, SAL, ESAH, ESAL, X2, Y2, D1);
  gram_gemm_kernel<<<32, 256, 0, stream>>>(ESA, C2, DDD);

  // Phase C: exact lower medians via 4-round byte radix select
  sel_init_kernel<<<16, 256, 0, stream>>>(ST1, ST2, HIST, SELF,
                                          8388607u /*(16M-1)/2*/, 32767u /*(64K-1)/2*/);
  const int shifts[4] = {24, 16, 8, 0};
  for (int r = 0; r < 4; ++r) {
    sel_count_kernel<<<4096, 256, 0, stream>>>(D1, N1 * N1, ST1, HIST, shifts[r]);
    sel_scan_kernel<<<1, 256, 0, stream>>>(ST1, HIST);
  }
  for (int r = 0; r < 4; ++r) {
    sel_count_kernel<<<256, 256, 0, stream>>>(DDD, 256 * 256, ST2, HIST, shifts[r]);
    sel_scan_kernel<<<1, 256, 0, stream>>>(ST2, HIST);
  }
  gamma_kernel<<<1, 32, 0, stream>>>(ST1, ST2, GAM);

  // Phase D: exp sums
  simil_kernel<<<N1, 256, 0, stream>>>(D1, GAM, SIM);
  self_gemm_kernel<<<512, 256, 0, stream>>>(SAH, SAL, X2, GAM, SELF);
  final_kernel<<<16, 256, 0, stream>>>(SIM, SELF, out);
}
